// SupConLoss_61040075211011
// MI455X (gfx1250) — compile-verified, matrix operand order checked by reading, not run on hardware
//
#include <hip/hip_runtime.h>
#include <hip/hip_bf16.h>

// ---------------------------------------------------------------------------
// SupConLoss, fused, for gfx1250 (MI455X).
//   N = 8192 rows (view-major), D = 128, BSZ = 512, VIEWS = 16.
//   GEMM (8192^2 x 128) on v_wmma_f32_16x16x32_f16; epilogue is
//   fma + v_exp_f32 + pk_add per element via:
//     S1_i = (1/T) * (x_i . P_{class(i)} - x_i.x_i)      (class pre-sums)
//     Z_i  = [sum_j exp((x_i.x_j - 1)/T)] - exp((x_i.x_i - 1)/T)  (diag fixup)
//   Row max is analytically 1/T for normalized rows.
//   LDS column tiles are filled with GLOBAL_LOAD_ASYNC_TO_LDS_B128 (ASYNCcnt).
// ---------------------------------------------------------------------------

typedef _Float16 v16h __attribute__((ext_vector_type(16)));
typedef _Float16 v8h  __attribute__((ext_vector_type(8)));
typedef _Float16 v4h  __attribute__((ext_vector_type(4)));
typedef float    v8f  __attribute__((ext_vector_type(8)));

#define SC_N      8192
#define SC_BSZ    512
#define SC_D      128
#define SC_INVT   14.285714285714286f     /* 1/0.07 */
#define SC_SCALE  20.609929155556625f     /* (1/0.07) * log2(e) */
#define SC_EPS    1e-12f
#define SC_NCLS   128                     /* labels are < 100; round up */

#define SPLITS    16                      /* column splits of 512 cols  */
#define ROWS_WG   256                     /* rows per workgroup (8 waves x 32) */
#define TILE_COLS 128
#define LDS_STRH  136   /* 128 + 8 halves pad: conflict-free ds_load_b128 */

union Frag16 { v16h v; v8h h[2]; };

// ---------------------------------------------------------------------------
// Kernel 1: L2-normalize each view, reorder to view-major, convert to f16.
// Output row n (= v*512 + b) comes from feature row b*16 + v. Wave per row.
// ---------------------------------------------------------------------------
__global__ void supcon_normalize_f16(const float* __restrict__ feat,
                                     _Float16* __restrict__ X) {
  const int n    = blockIdx.x * 8 + (threadIdx.x >> 5);
  const int lane = threadIdx.x & 31;
  const int b = n & (SC_BSZ - 1);
  const int v = n >> 9;
  const float4 x = *(const float4*)(feat + (size_t)(b * 16 + v) * SC_D + lane * 4);
  float ss = x.x * x.x + x.y * x.y + x.z * x.z + x.w * x.w;
#pragma unroll
  for (int m = 1; m < 32; m <<= 1) ss += __shfl_xor(ss, m, 32);
  const float inv = 1.0f / fmaxf(sqrtf(ss), SC_EPS);
  v4h o;
  o[0] = (_Float16)(x.x * inv);
  o[1] = (_Float16)(x.y * inv);
  o[2] = (_Float16)(x.z * inv);
  o[3] = (_Float16)(x.w * inv);
  *(v4h*)(X + (size_t)n * SC_D + lane * 4) = o;
}

// ---------------------------------------------------------------------------
// Kernel 2: positive-pair counts per sample: C[b] = 16*count(labels==lab_b)-1
// ---------------------------------------------------------------------------
__global__ void supcon_crow(const int* __restrict__ labels,
                            float* __restrict__ Crow) {
  __shared__ int sl[SC_BSZ];
  const int t = threadIdx.x;          // blockDim.x == 512
  sl[t] = labels[t];
  __syncthreads();
  const int my = sl[t];
  int cnt = 0;
  for (int j = 0; j < SC_BSZ; ++j) cnt += (sl[j] == my) ? 1 : 0;
  Crow[t] = (float)(16 * cnt - 1);
}

// ---------------------------------------------------------------------------
// Kernel 3: class feature sums P[c][k] = sum over rows with label c of x[k].
// Grid 64 x 256: thread -> (class c, dim k).
// ---------------------------------------------------------------------------
__global__ void supcon_psum(const _Float16* __restrict__ X,
                            const int* __restrict__ labels,
                            float* __restrict__ P) {
  __shared__ int sl[SC_BSZ];
  const int tid = threadIdx.x;        // 256
  sl[tid]       = labels[tid];
  sl[tid + 256] = labels[tid + 256];
  __syncthreads();
  const int t = blockIdx.x * 256 + tid;     // 0 .. 16383
  const int c = t >> 7;
  const int k = t & (SC_D - 1);
  float acc = 0.f;
  for (int b = 0; b < SC_BSZ; ++b) {
    if (sl[b] == c) {
#pragma unroll
      for (int v = 0; v < 16; ++v)
        acc += (float)X[(size_t)(v * SC_BSZ + b) * SC_D + k];
    }
  }
  P[c * SC_D + k] = acc;
}

// ---------------------------------------------------------------------------
// Kernel 4: fused similarity / exp-sum pass.
// Block: 256 threads (8 waves). Wave w owns 32 rows (two 16-row A tiles,
// fragments register-resident). Grid: (32 row tiles, 16 column splits of 512).
// 128x128 f16 column tiles DMA'd into padded LDS via async load-to-LDS;
// each B fragment feeds 2 WMMAs. Per-element epilogue: fma + exp + add.
// ---------------------------------------------------------------------------
__global__ void __launch_bounds__(256)
supcon_main(const _Float16* __restrict__ X,
            float* __restrict__ Zpart) {
  __shared__ _Float16 sB[TILE_COLS * LDS_STRH];

  const int tid    = threadIdx.x;
  const int waveId = tid >> 5;
  const int lane   = tid & 31;
  const int l16    = lane & 15;
  const int halfId = lane >> 4;

  const int split       = blockIdx.y;                      // 0..15
  const int splitBase   = split * (SC_N / SPLITS);         // *512
  const int waveRowBase = blockIdx.x * ROWS_WG + waveId * 32;
  const int row0_t0     = waveRowBase + 8 * halfId;        // zacc0 rows
  const int row0_t1     = waveRowBase + 16 + 8 * halfId;   // zacc1 rows

  // A fragments: two 16-row tiles x 4 K-chunks, register resident.
  Frag16 a0[4], a1[4];
  {
    const _Float16* p0 = X + (size_t)(waveRowBase + l16) * SC_D;
    const _Float16* p1 = X + (size_t)(waveRowBase + 16 + l16) * SC_D;
#pragma unroll
    for (int c = 0; c < 4; ++c) {
      a0[c].h[0] = *(const v8h*)(p0 + 32 * c + 8 * halfId);
      a0[c].h[1] = *(const v8h*)(p0 + 32 * c + 16 + 8 * halfId);
      a1[c].h[0] = *(const v8h*)(p1 + 32 * c + 8 * halfId);
      a1[c].h[1] = *(const v8h*)(p1 + 32 * c + 16 + 8 * halfId);
    }
  }

  float zacc0[8], zacc1[8];
#pragma unroll
  for (int v = 0; v < 8; ++v) { zacc0[v] = 0.f; zacc1[v] = 0.f; }

  const int ITERS = (SC_N / SPLITS) / TILE_COLS;           // 4
  for (int it = 0; it < ITERS; ++it) {
    const int colBase = splitBase + it * TILE_COLS;

    __syncthreads();  // previous tile fully consumed by all waves
    // Async DMA fill (no VGPR data): 128 rows x 256 B, 16B chunks, 8/thread.
    // GVS addressing: mem = s[base] + v[offset]; LDS dest = VGPR byte addr.
#pragma unroll
    for (int i = 0; i < 8; ++i) {
      const int cid = tid + i * 256;       // 0..2047
      const int r   = cid >> 4;
      const int kc  = cid & 15;
      const unsigned ldsOff =
          (unsigned)(size_t)(&sB[r * LDS_STRH + kc * 8]);
      const unsigned gOff = (unsigned)((colBase + r) * (SC_D * 2) + kc * 16);
      asm volatile("global_load_async_to_lds_b128 %0, %1, %2"
                   :
                   : "v"(ldsOff), "v"(gOff), "s"(X)
                   : "memory");
    }
    if (it < ITERS - 1) {  // warm L2 for the next tile (global_prefetch_b8)
      __builtin_prefetch(X + (size_t)(colBase + TILE_COLS + (tid >> 1)) * SC_D +
                             (tid & 1) * 64, 0, 1);
    }
    asm volatile("s_wait_asynccnt 0" ::: "memory");
    __syncthreads();

#pragma unroll
    for (int sub = 0; sub < 8; ++sub) {
      const _Float16* bp = &sB[(sub * 16 + l16) * LDS_STRH];
      // Hoist all four B-chunk fragments: loads issue back-to-back so the
      // scheduler can use partial dscnt waits and overlap with WMMA.
      Frag16 bfr[4];
#pragma unroll
      for (int c = 0; c < 4; ++c) {
        bfr[c].h[0] = *(const v8h*)(bp + 32 * c + 8 * halfId);
        bfr[c].h[1] = *(const v8h*)(bp + 32 * c + 16 + 8 * halfId);
      }
      v8f acc0 = {};
      v8f acc1 = {};
#pragma unroll
      for (int c = 0; c < 4; ++c) {
        acc0 = __builtin_amdgcn_wmma_f32_16x16x32_f16(
            false, a0[c].v, false, bfr[c].v, (short)0, acc0, false, false);
        acc1 = __builtin_amdgcn_wmma_f32_16x16x32_f16(
            false, a1[c].v, false, bfr[c].v, (short)0, acc1, false, false);
      }
      // epilogue: exp((sim-1)/T) = exp2(acc*S - S); independent temp chains
      float e0[8], e1[8];
#pragma unroll
      for (int v = 0; v < 8; ++v) {
        e0[v] = __builtin_amdgcn_exp2f(
            __builtin_fmaf(acc0[v], SC_SCALE, -SC_SCALE));
        e1[v] = __builtin_amdgcn_exp2f(
            __builtin_fmaf(acc1[v], SC_SCALE, -SC_SCALE));
      }
#pragma unroll
      for (int v = 0; v < 8; ++v) {
        zacc0[v] += e0[v];
        zacc1[v] += e1[v];
      }
    }
  }

  // reduce across the 16 lanes of each half (xor 1,2,4,8 stay within halves)
#pragma unroll
  for (int v = 0; v < 8; ++v) {
#pragma unroll
    for (int m = 1; m < 16; m <<= 1) {
      zacc0[v] += __shfl_xor(zacc0[v], m, 32);
      zacc1[v] += __shfl_xor(zacc1[v], m, 32);
    }
  }
  if (l16 == 0) {
    float* zdst = Zpart + (size_t)split * SC_N;
#pragma unroll
    for (int v = 0; v < 8; ++v) {
      zdst[row0_t0 + v] = zacc0[v];
      zdst[row0_t1 + v] = zacc1[v];
    }
  }
}

// ---------------------------------------------------------------------------
// Kernel 5: per-row loss. Wave per row: dot(x_i, P_class), |x_i|^2, combine
// Z partials, diagonal fixup, closed-form loss.
//   loss_i = -[ (1/T)(x.P - ssq) - C*(1/T + log(Z+eps)) ] / (C + eps)
// ---------------------------------------------------------------------------
__global__ void supcon_rowloss(const _Float16* __restrict__ X,
                               const int* __restrict__ labels,
                               const float* __restrict__ Zpart,
                               const float* __restrict__ Crow,
                               const float* __restrict__ P,
                               float* __restrict__ rowloss) {
  const int n    = blockIdx.x * 8 + (threadIdx.x >> 5);
  const int lane = threadIdx.x & 31;
  const int b    = n & (SC_BSZ - 1);
  const int c    = labels[b];

  const v4h xh = *(const v4h*)(X + (size_t)n * SC_D + lane * 4);
  const float* Pc = P + c * SC_D + lane * 4;
  float ssq = 0.f, dp = 0.f;
#pragma unroll
  for (int j = 0; j < 4; ++j) {
    const float xv = (float)xh[j];
    ssq += xv * xv;
    dp  += xv * Pc[j];
  }
#pragma unroll
  for (int m = 1; m < 32; m <<= 1) {
    ssq += __shfl_xor(ssq, m, 32);
    dp  += __shfl_xor(dp, m, 32);
  }
  if (lane == 0) {
    float Z = 0.f;
#pragma unroll
    for (int s = 0; s < SPLITS; ++s) Z += Zpart[(size_t)s * SC_N + n];
    Z -= __builtin_amdgcn_exp2f((ssq - 1.f) * SC_SCALE);   // drop diagonal
    const float S1   = SC_INVT * (dp - ssq);
    const float C    = Crow[b];
    const float mlpp = (S1 - C * (SC_INVT + __logf(Z + SC_EPS))) / (C + SC_EPS);
    rowloss[n] = -mlpp;
  }
}

// ---------------------------------------------------------------------------
// Kernel 6: mean over 8192 per-row losses -> scalar.
// ---------------------------------------------------------------------------
__global__ void supcon_reduce(const float* __restrict__ rowloss,
                              float* __restrict__ out) {
  __shared__ float red[256];
  const int tid = threadIdx.x;
  float acc = 0.f;
  for (int r = tid; r < SC_N; r += 256) acc += rowloss[r];
  red[tid] = acc;
  __syncthreads();
  for (int s = 128; s > 0; s >>= 1) {
    if (tid < s) red[tid] += red[tid + s];
    __syncthreads();
  }
  if (tid == 0) out[0] = red[0] * (1.0f / (float)SC_N);
}

// ---------------------------------------------------------------------------
extern "C" void kernel_launch(void* const* d_in, const int* in_sizes, int n_in,
                              void* d_out, int out_size, void* d_ws, size_t ws_size,
                              hipStream_t stream) {
  (void)in_sizes; (void)n_in; (void)out_size; (void)ws_size;
  const float* feat   = (const float*)d_in[0];
  const int*   labels = (const int*)d_in[1];
  float*       out    = (float*)d_out;

  char* ws = (char*)d_ws;
  _Float16* X    = (_Float16*)ws;                            // 2 MiB
  float*    Zp   = (float*)(ws + (size_t)2097152);           // 16*8192*4 = 512 KiB
  float*    Crow = (float*)(ws + (size_t)2621440);           // 2 KiB
  float*    P    = (float*)(ws + (size_t)2625536);           // 128*128*4 = 64 KiB
  float*    Rl   = (float*)(ws + (size_t)2691072);           // 8192*4 = 32 KiB

  supcon_normalize_f16<<<SC_N / 8, 256, 0, stream>>>(feat, X);
  supcon_crow<<<1, SC_BSZ, 0, stream>>>(labels, Crow);
  supcon_psum<<<(SC_NCLS * SC_D) / 256, 256, 0, stream>>>(X, labels, P);
  supcon_main<<<dim3(SC_N / ROWS_WG, SPLITS), 256, 0, stream>>>(X, Zp);
  supcon_rowloss<<<SC_N / 8, 256, 0, stream>>>(X, labels, Zp, Crow, P, Rl);
  supcon_reduce<<<1, 256, 0, stream>>>(Rl, out);
}
